// TransformerXLHybridEncoder_16484084482890
// MI455X (gfx1250) — matrix operand order repaired and checked by reading, not compiled, not run on hardware
//
#include <hip/hip_runtime.h>

// ---------------- problem constants ----------------
constexpr int QLEN = 1024;
constexpr int MLEN = 1024;
constexpr int KLEN = 2048;   // QLEN + MLEN
constexpr int BSZ  = 4;
constexpr int DM   = 1024;   // d_model
constexpr int NH   = 16;     // n_head
constexpr int DH   = 64;     // d_head
constexpr int HDM  = 1024;   // NH*DH
constexpr int QKV  = 3072;   // 3*HDM
constexpr float SCALE = 0.125f;  // 1/sqrt(64)

// ---------------- WMMA types ----------------
typedef __attribute__((ext_vector_type(16))) _Float16 v16h;
typedef __attribute__((ext_vector_type(8)))  _Float16 v8h;
typedef __attribute__((ext_vector_type(8)))  float    v8f;

union V16 { v16h v; v8h h[2]; };

__device__ inline v8f wmma_f16(v16h a, v16h b, v8f c) {
  // D = A(16x32 f16) x B(32x16 f16) + C(16x16 f32)
  return __builtin_amdgcn_wmma_f32_16x16x32_f16(false, a, false, b, (short)0, c,
                                                false, false);
}

__device__ inline void wave_lds_sync() {
  __builtin_amdgcn_wave_barrier();
  asm volatile("s_wait_dscnt 0" ::: "memory");
  __builtin_amdgcn_wave_barrier();
}

// ---------------- setup / conversion kernels ----------------
__global__ void k_build_cat(const float* __restrict__ mems,
                            const float* __restrict__ w,
                            _Float16* __restrict__ cat) {
  size_t i = (size_t)blockIdx.x * 256 + threadIdx.x;   // 8192*1024 elements
  const size_t memsN = (size_t)MLEN * BSZ * DM;
  float v = (i < memsN) ? mems[i] : w[i - memsN];
  cat[i] = (_Float16)v;
}

__global__ void k_convert(const float* __restrict__ src,
                          _Float16* __restrict__ dst, size_t n) {
  size_t i = (size_t)blockIdx.x * 256 + threadIdx.x;
  if (i < n) dst[i] = (_Float16)src[i];
}

__global__ void k_transpose(const float* __restrict__ src,   // [K x N]
                            _Float16* __restrict__ dst,      // [N x K]
                            int K, int N) {
  size_t i = (size_t)blockIdx.x * 256 + threadIdx.x;
  if (i >= (size_t)K * N) return;
  int k = (int)(i / N);
  int n = (int)(i % N);
  dst[(size_t)n * K + k] = (_Float16)src[i];
}

// vT[b][n][d][key] = heads[key*BSZ+b][2*HDM + n*DH + d]
__global__ void k_build_vt(const _Float16* __restrict__ heads,
                           _Float16* __restrict__ vt) {
  size_t i = (size_t)blockIdx.x * 256 + threadIdx.x;    // 4*16*64*2048
  int key = (int)(i & (KLEN - 1));
  size_t r = i >> 11;
  int d = (int)(r & (DH - 1));
  size_t bn = r >> 6;
  int n = (int)(bn & (NH - 1));
  int b = (int)(bn >> 4);
  vt[i] = heads[((size_t)key * BSZ + b) * QKV + 2 * HDM + n * DH + d];
}

// ---------------- generic WMMA GEMM: C[M,N] = A[M,K] * BT[N,K]^T ----------------
template <bool F16OUT>
__global__ __launch_bounds__(128) void gemm16(const _Float16* __restrict__ A,
                                              const _Float16* __restrict__ BT,
                                              void* __restrict__ Cout,
                                              int M, int N, int K) {
  const int lane = threadIdx.x & 31;
  const int wv   = threadIdx.x >> 5;      // 4 waves stacked along M
  const int lh   = lane & 15;
  const int hi   = lane >> 4;             // half-wave id
  const int m0 = blockIdx.y * 64 + wv * 16;
  const int n0 = blockIdx.x * 64;

  v8f acc[4] = {};
  const _Float16* arow = A + (size_t)(m0 + lh) * K;
  const _Float16* brow0 = BT + (size_t)(n0 + 0 * 16 + lh) * K;
  const _Float16* brow1 = BT + (size_t)(n0 + 1 * 16 + lh) * K;
  const _Float16* brow2 = BT + (size_t)(n0 + 2 * 16 + lh) * K;
  const _Float16* brow3 = BT + (size_t)(n0 + 3 * 16 + lh) * K;
  const _Float16* brow[4] = {brow0, brow1, brow2, brow3};

  for (int kk = 0; kk < K; kk += 32) {
    V16 af;
    af.h[0] = *(const v8h*)(arow + kk + hi * 16);        // K = kk + hi*16 .. +7
    af.h[1] = *(const v8h*)(arow + kk + hi * 16 + 8);    // K = kk + hi*16+8 ..
#pragma unroll
    for (int nt = 0; nt < 4; ++nt) {
      V16 bf;
      bf.h[0] = *(const v8h*)(brow[nt] + kk + hi * 16);      // K = kk+hi*16..+7
      bf.h[1] = *(const v8h*)(brow[nt] + kk + hi * 16 + 8);  // ..+15
      acc[nt] = wmma_f16(af.v, bf.v, acc[nt]);
    }
  }
#pragma unroll
  for (int nt = 0; nt < 4; ++nt) {
#pragma unroll
    for (int v = 0; v < 8; ++v) {
      const size_t row = (size_t)(m0 + hi * 8 + v);
      const int col = n0 + nt * 16 + lh;
      if (F16OUT)
        ((_Float16*)Cout)[row * (size_t)N + col] = (_Float16)acc[nt][v];
      else
        ((float*)Cout)[row * (size_t)N + col] = acc[nt][v];
    }
  }
}

// ---------------- fused rel-attention (flash-style online softmax) ----------------
// grid: (16 q-groups, NH, BSZ), block 128 (4 waves; wave wv owns q-tile q0)
__global__ __launch_bounds__(128) void attn_kernel(
    const _Float16* __restrict__ heads,   // [KLEN*BSZ, 3072] (q|k|v)
    const _Float16* __restrict__ rh,      // [KLEN, HDM]
    const _Float16* __restrict__ vt,      // [BSZ,NH,DH,KLEN]
    const float* __restrict__ rwb,        // r_w_bias [NH,DH]
    const float* __restrict__ rrb,        // r_r_bias [NH,DH]
    _Float16* __restrict__ attnvec)       // [QLEN*BSZ, HDM]
{
  __shared__ __align__(16) float    bd_s[4][16][48];
  __shared__ __align__(16) _Float16 p_s[4][16][32];

  const int lane = threadIdx.x & 31;
  const int wv   = threadIdx.x >> 5;
  const int lh   = lane & 15;
  const int hi   = lane >> 4;
  const int q0 = (blockIdx.x * 4 + wv) * 16;
  const int n  = blockIdx.y;
  const int b  = blockIdx.z;

  // Q fragments with +r_w_bias / +r_r_bias, K-dim = d_head split into 2x32
  const _Float16* qrow =
      heads + ((size_t)(MLEN + q0 + lh) * BSZ + b) * QKV + n * DH;
  v16h qw[2], qr[2];
#pragma unroll
  for (int f = 0; f < 2; ++f) {
    const int k0 = f * 32 + hi * 16;
    v8h lo  = *(const v8h*)(qrow + k0);
    v8h hi8 = *(const v8h*)(qrow + k0 + 8);
    V16 aw, ar;
#pragma unroll
    for (int ii = 0; ii < 8; ++ii) {
      float ql = (float)lo[ii], qh = (float)hi8[ii];
      aw.h[0][ii] = (_Float16)(ql + rwb[n * DH + k0 + ii]);
      aw.h[1][ii] = (_Float16)(qh + rwb[n * DH + k0 + 8 + ii]);
      ar.h[0][ii] = (_Float16)(ql + rrb[n * DH + k0 + ii]);
      ar.h[1][ii] = (_Float16)(qh + rrb[n * DH + k0 + 8 + ii]);
    }
    qw[f] = aw.v;
    qr[f] = ar.v;
  }

  v8f O[4] = {};
  float mrow[8], lrow[8];
#pragma unroll
  for (int v = 0; v < 8; ++v) { mrow[v] = -3.0e38f; lrow[v] = 0.0f; }

  const int nchunk = (q0 + 15 + MLEN) / 32 + 1;   // last visible key tile
  for (int ch = 0; ch < nchunk; ++ch) {
    const int j0 = ch * 32;

    // ---- AC = (q + r_w_bias) . k  for two 16-key tiles ----
    v8f sAC[2];
#pragma unroll
    for (int t = 0; t < 2; ++t) {
      const int j = j0 + t * 16 + lh;               // j < KLEN always
      const _Float16* krow =
          heads + ((size_t)j * BSZ + b) * QKV + HDM + n * DH + hi * 16;
      V16 b0, b1;
      b0.h[0] = *(const v8h*)(krow);        b0.h[1] = *(const v8h*)(krow + 8);
      b1.h[0] = *(const v8h*)(krow + 32);   b1.h[1] = *(const v8h*)(krow + 40);
      v8f acc = {};
      acc = wmma_f16(qw[0], b0.v, acc);
      acc = wmma_f16(qw[1], b1.v, acc);
      sAC[t] = acc;
    }

    // ---- BD_orig tiles covering the rel-shift diagonal band ----
    // BD_shifted[i,j] = BD_orig[i, j-i+QLEN-1]; band base C0 is 16-aligned.
    const int C0 = j0 - q0 + (QLEN - 16);
#pragma unroll
    for (int u = 0; u < 3; ++u) {
      int c = C0 + u * 16 + lh;
      if (c > KLEN - 1) c = KLEN - 1;               // clamped cols are masked
      const _Float16* rrow = rh + (size_t)c * HDM + n * DH + hi * 16;
      V16 b0, b1;
      b0.h[0] = *(const v8h*)(rrow);        b0.h[1] = *(const v8h*)(rrow + 8);
      b1.h[0] = *(const v8h*)(rrow + 32);   b1.h[1] = *(const v8h*)(rrow + 40);
      v8f acc = {};
      acc = wmma_f16(qr[0], b0.v, acc);
      acc = wmma_f16(qr[1], b1.v, acc);
#pragma unroll
      for (int v = 0; v < 8; ++v)
        bd_s[wv][v + hi * 8][u * 16 + lh] = acc[v];
    }
    wave_lds_sync();

    // ---- scores, mask, online softmax ----
#pragma unroll
    for (int v = 0; v < 8; ++v) {
      const int M = v + hi * 8;
      const int i = q0 + M;
      int j = j0 + lh;
      float s0 = (sAC[0][v] + bd_s[wv][M][lh - M + 15]) * SCALE;
      if (j > i + MLEN) s0 = -3.0e38f;
      j = j0 + 16 + lh;
      float s1 = (sAC[1][v] + bd_s[wv][M][16 + lh - M + 15]) * SCALE;
      if (j > i + MLEN) s1 = -3.0e38f;

      float mx = fmaxf(s0, s1);
#pragma unroll
      for (int off = 1; off < 16; off <<= 1)
        mx = fmaxf(mx, __shfl_xor(mx, off, 32));    // stays in 16-lane half
      const float mnew  = fmaxf(mrow[v], mx);
      const float alpha = __expf(mrow[v] - mnew);
      const float e0 = __expf(s0 - mnew);
      const float e1 = __expf(s1 - mnew);
      float ps = e0 + e1;
#pragma unroll
      for (int off = 1; off < 16; off <<= 1) ps += __shfl_xor(ps, off, 32);
      lrow[v] = lrow[v] * alpha + ps;
      mrow[v] = mnew;
#pragma unroll
      for (int nt = 0; nt < 4; ++nt) O[nt][v] *= alpha;
      p_s[wv][M][lh]      = (_Float16)e0;
      p_s[wv][M][16 + lh] = (_Float16)e1;
    }
    wave_lds_sync();

    // ---- O += P(16x32) x V(32x64) ----
    V16 pa;
    pa.h[0] = *(const v8h*)&p_s[wv][lh][hi * 16];
    pa.h[1] = *(const v8h*)&p_s[wv][lh][hi * 16 + 8];
#pragma unroll
    for (int nt = 0; nt < 4; ++nt) {
      const int d = nt * 16 + lh;
      const _Float16* vrow =
          vt + ((size_t)(b * NH + n) * DH + d) * KLEN + j0 + hi * 16;
      V16 vb;
      vb.h[0] = *(const v8h*)(vrow);
      vb.h[1] = *(const v8h*)(vrow + 8);
      O[nt] = wmma_f16(pa.v, vb.v, O[nt]);
    }
    wave_lds_sync();
  }

  // ---- finalize: O / l -> attn_vec (f16, rows (i*BSZ+b), col n*DH+d) ----
#pragma unroll
  for (int v = 0; v < 8; ++v) {
    const int i = q0 + v + hi * 8;
    const float inv = 1.0f / lrow[v];
#pragma unroll
    for (int nt = 0; nt < 4; ++nt) {
      const int col = n * DH + nt * 16 + lh;
      attnvec[((size_t)i * BSZ + b) * HDM + col] = (_Float16)(O[nt][v] * inv);
    }
  }
}

// ---------------- residual + LayerNorm ----------------
__global__ __launch_bounds__(256) void ln_kernel(
    const float* __restrict__ w, const float* __restrict__ attnout,
    const float* __restrict__ gamma, const float* __restrict__ beta,
    float* __restrict__ out) {
  const int row = blockIdx.x;
  const float* xw = w + (size_t)row * DM;
  const float* xa = attnout + (size_t)row * DM;
  float xs[4], s = 0.f, s2 = 0.f;
#pragma unroll
  for (int t = 0; t < 4; ++t) {
    const int c = threadIdx.x + t * 256;
    float x = xw[c] + xa[c];
    xs[t] = x; s += x; s2 += x * x;
  }
  __shared__ float rs[256], rs2[256];
  rs[threadIdx.x] = s; rs2[threadIdx.x] = s2;
  __syncthreads();
  for (int off = 128; off > 0; off >>= 1) {
    if (threadIdx.x < off) {
      rs[threadIdx.x]  += rs[threadIdx.x + off];
      rs2[threadIdx.x] += rs2[threadIdx.x + off];
    }
    __syncthreads();
  }
  const float mean = rs[0] / DM;
  const float var  = rs2[0] / DM - mean * mean;
  const float rstd = rsqrtf(var + 1e-5f);
#pragma unroll
  for (int t = 0; t < 4; ++t) {
    const int c = threadIdx.x + t * 256;
    out[(size_t)row * DM + c] = (xs[t] - mean) * rstd * gamma[c] + beta[c];
  }
}

// ---------------- host launcher ----------------
extern "C" void kernel_launch(void* const* d_in, const int* in_sizes, int n_in,
                              void* d_out, int out_size, void* d_ws,
                              size_t ws_size, hipStream_t stream) {
  (void)in_sizes; (void)n_in; (void)out_size; (void)ws_size;
  const float* w      = (const float*)d_in[0];
  const float* r      = (const float*)d_in[1];
  const float* mems   = (const float*)d_in[2];
  const float* W_qkv  = (const float*)d_in[3];
  const float* W_r    = (const float*)d_in[4];
  const float* W_o    = (const float*)d_in[5];
  const float* rwb    = (const float*)d_in[6];
  const float* rrb    = (const float*)d_in[7];
  const float* gamma  = (const float*)d_in[8];
  const float* beta   = (const float*)d_in[9];
  float* out = (float*)d_out;

  // workspace bump allocation (~128 MB total)
  char* p = (char*)d_ws;
  auto alloc = [&](size_t bytes) -> char* {
    char* ret = p;
    p += (bytes + 255) & ~(size_t)255;
    return ret;
  };
  _Float16* catF16  = (_Float16*)alloc((size_t)KLEN * BSZ * DM * 2);   // 16 MB
  _Float16* wqkvT   = (_Float16*)alloc((size_t)QKV * DM * 2);          //  6 MB
  _Float16* rF16    = (_Float16*)alloc((size_t)KLEN * DM * 2);         //  4 MB
  _Float16* wrT     = (_Float16*)alloc((size_t)HDM * DM * 2);          //  2 MB
  _Float16* woT     = (_Float16*)alloc((size_t)DM * HDM * 2);          //  2 MB
  _Float16* headsF  = (_Float16*)alloc((size_t)KLEN * BSZ * QKV * 2);  // 48 MB
  _Float16* rhF     = (_Float16*)alloc((size_t)KLEN * HDM * 2);        //  4 MB
  _Float16* vT      = (_Float16*)alloc((size_t)BSZ * NH * DH * KLEN * 2); // 16 MB
  _Float16* attnvec = (_Float16*)alloc((size_t)QLEN * BSZ * HDM * 2);  //  8 MB
  float*    attnout = (float*)alloc((size_t)QLEN * BSZ * DM * 4);      // 16 MB

  // 1) f16 conversions / transposes
  k_build_cat<<<(KLEN * BSZ * DM) / 256, 256, 0, stream>>>(mems, w, catF16);
  k_transpose<<<(DM * QKV) / 256, 256, 0, stream>>>(W_qkv, wqkvT, DM, QKV);
  k_convert<<<(KLEN * DM) / 256, 256, 0, stream>>>(r, rF16, (size_t)KLEN * DM);
  k_transpose<<<(DM * HDM) / 256, 256, 0, stream>>>(W_r, wrT, DM, HDM);
  k_transpose<<<(HDM * DM) / 256, 256, 0, stream>>>(W_o, woT, HDM, DM);

  // 2) QKV projection: heads[8192,3072] = cat[8192,1024] x W_qkv
  gemm16<true><<<dim3(QKV / 64, (KLEN * BSZ) / 64), 128, 0, stream>>>(
      catF16, wqkvT, headsF, KLEN * BSZ, QKV, DM);

  // 3) position projection: rh[2048,1024] = r x W_r
  gemm16<true><<<dim3(HDM / 64, KLEN / 64), 128, 0, stream>>>(
      rF16, wrT, rhF, KLEN, HDM, DM);

  // 4) V transpose for PV B-fragments
  k_build_vt<<<(BSZ * NH * DH * KLEN) / 256, 256, 0, stream>>>(headsF, vT);

  // 5) fused relative attention
  attn_kernel<<<dim3(QLEN / 64, NH, BSZ), 128, 0, stream>>>(
      headsF, rhF, vT, rwb, rrb, attnvec);

  // 6) output projection: attnout[4096,1024] = attnvec x W_o
  gemm16<false><<<dim3(DM / 64, (QLEN * BSZ) / 64), 128, 0, stream>>>(
      attnvec, woT, attnout, QLEN * BSZ, DM, HDM);

  // 7) residual + LayerNorm
  ln_kernel<<<QLEN * BSZ, 256, 0, stream>>>(w, attnout, gamma, beta, out);
}
